// ActorCriticRNN_75076028334729
// MI455X (gfx1250) — compile-verified, matrix operand order checked by reading, not run on hardware
//
#include <hip/hip_runtime.h>

#define TT   64
#define NB   256
#define HIMG 6
#define WIMG 9
#define CIN  26
#define HWP  54
#define TN   (TT*NB)      // 16384
#define M1   (TN*HWP)     // 884736
#define HID  128

// LDS row strides padded for conflict-free ds_load_b128 (stride/2 DWORDs, gcd with 64 == 4)
#define SX   40           // X tile (K=32)
#define SH   136          // 128-wide activation tiles
#define KP4  104          // conv4 im2col stride (K=96)
#define KP5  168          // conv5 im2col stride (K=160)
#define KP6  296          // conv6 im2col stride (K=288)

typedef __attribute__((ext_vector_type(16))) __bf16         v16bf;
typedef __attribute__((ext_vector_type(2)))  __bf16         v2bf;
typedef __attribute__((ext_vector_type(16))) unsigned short v16u;
typedef __attribute__((ext_vector_type(8)))  unsigned short v8u;
typedef __attribute__((ext_vector_type(8)))  float          v8f;
typedef __attribute__((ext_vector_type(2)))  float          v2f;

// native f32 -> bf16 (RNE) via fptrunc; lowers to v_cvt_pk_bf16_f32 on gfx1250
__device__ __forceinline__ unsigned short f2bf(float x){
  __bf16 b = (__bf16)x;
  return __builtin_bit_cast(unsigned short, b);
}
// packed 2x f32 -> 2x bf16
__device__ __forceinline__ unsigned int f2bf_pk(float lo, float hi){
  v2f f; f[0] = lo; f[1] = hi;
  v2bf b = __builtin_convertvector(f, v2bf);
  return __builtin_bit_cast(unsigned int, b);
}
__device__ __forceinline__ float relu_(float x){ return x > 0.f ? x : 0.f; }
__device__ __forceinline__ float sigm_(float x){ return 1.f/(1.f+__expf(-x)); }
__device__ __forceinline__ v8f zero8(){
  v8f z;
  #pragma unroll
  for (int i=0;i<8;i++) z[i]=0.f;
  return z;
}

__device__ __forceinline__ v8f wmma_bf16(v16u a, v16u b, v8f c){
  return __builtin_amdgcn_wmma_f32_16x16x32_bf16(
      false, __builtin_bit_cast(v16bf, a),
      false, __builtin_bit_cast(v16bf, b),
      (short)0, c, false, false);
}

// A/B fragment for 16x32 bf16: lane l holds row (l&15); K-halves split by l>>4.
// base is row-major (stride halfs, multiple of 8 -> 16B aligned chunks).
__device__ __forceinline__ v16u frag_u16(const unsigned short* base, int stride, int kbase){
  const int lane = threadIdx.x & 31;
  const unsigned short* p = base + (lane & 15) * stride + kbase + ((lane >> 4) << 3);
  v8u lo = *(const v8u*)p;
  v8u hi = *(const v8u*)(p + 16);
  v16u f;
  #pragma unroll
  for (int j=0;j<8;j++){ f[j]=lo[j]; f[j+8]=hi[j]; }
  return f;
}

// acc += A(16xK, strideA) * Wt tile ct (Wt packed [col][k], stride Kpad)
__device__ __forceinline__ v8f gemm_tile(const unsigned short* A, int strideA,
                                         const unsigned short* Wt, int Kpad,
                                         int ct, int K, v8f acc){
  const unsigned short* B = Wt + ct * 16 * Kpad;
  for (int k=0; k<K; k+=32)
    acc = wmma_bf16(frag_u16(A, strideA, k), frag_u16(B, Kpad, k), acc);
  return acc;
}

// store a D tile (v8f) as relu(x+bias) in bf16 to row-major buffer [row][col], stride halfs
__device__ __forceinline__ void store_tile_relu_bf16(unsigned short* dst, int stride, int col,
                                                     v8f acc, float bias, int hi8){
  #pragma unroll
  for (int i=0;i<8;i+=2){
    unsigned int p = f2bf_pk(relu_(acc[i] + bias), relu_(acc[i+1] + bias));
    dst[(i + hi8)*stride + col]     = (unsigned short)(p & 0xFFFFu);
    dst[(i + 1 + hi8)*stride + col] = (unsigned short)(p >> 16);
  }
}

// ---------------- weight pack: src (Ksrc x Nsrc) f32 row-major -> dst [col][k] bf16, zero padded
__global__ void k_pack(const float* __restrict__ src, unsigned short* __restrict__ dst,
                       int Ksrc, int Nsrc, int Kpad, int Npad){
  int idx = blockIdx.x * 256 + threadIdx.x;
  if (idx >= Kpad * Npad) return;
  int col = idx / Kpad, k = idx - col * Kpad;
  float v = (k < Ksrc && col < Nsrc) ? src[k * Nsrc + col] : 0.f;
  dst[idx] = f2bf(v);
}

// ---------------- fused 1x1 conv stack: relu(relu(relu(X@W1)@W2)@W3), per-pixel rows
__global__ void k_conv1x1(const float* __restrict__ obs,
                          const unsigned short* __restrict__ wt1, const float* __restrict__ b1,
                          const unsigned short* __restrict__ wt2, const float* __restrict__ b2,
                          const unsigned short* __restrict__ wt3, const float* __restrict__ b3,
                          unsigned short* __restrict__ c3out){
  extern __shared__ unsigned short smem[];
  const int wave = threadIdx.x >> 5, lane = threadIdx.x & 31;
  unsigned short* X  = smem + wave * (16*SX + 2*16*SH);
  unsigned short* Y1 = X + 16*SX;
  unsigned short* Y2 = Y1 + 16*SH;
  const long long rowbase = (long long)blockIdx.x * 128 + wave * 16;
  for (int e = lane; e < 512; e += 32){           // 16 rows x 32 (26 valid K)
    int r = e >> 5, k = e & 31;
    float v = (k < CIN) ? obs[(rowbase + r) * CIN + k] : 0.f;
    X[r*SX + k] = f2bf(v);
  }
  const int colb = lane & 15, hi8 = (lane >> 4) << 3;
  #pragma unroll
  for (int ct=0; ct<8; ct++){                     // conv1: K=32
    v8f acc = gemm_tile(X, SX, wt1, 32, ct, 32, zero8());
    const int col = ct*16 + colb;
    store_tile_relu_bf16(Y1, SH, col, acc, b1[col], hi8);
  }
  #pragma unroll
  for (int ct=0; ct<8; ct++){                     // conv2: K=128
    v8f acc = gemm_tile(Y1, SH, wt2, 128, ct, 128, zero8());
    const int col = ct*16 + colb;
    store_tile_relu_bf16(Y2, SH, col, acc, b2[col], hi8);
  }
  {                                               // conv3: K=128, 8 real cols
    v8f acc = gemm_tile(Y2, SH, wt3, 128, 0, 128, zero8());
    if (colb < 8){
      const float bias = b3[colb];
      #pragma unroll
      for (int i=0;i<8;i+=2){
        unsigned int p = f2bf_pk(relu_(acc[i] + bias), relu_(acc[i+1] + bias));
        c3out[(rowbase + i + hi8) * 8 + colb]     = (unsigned short)(p & 0xFFFFu);
        c3out[(rowbase + i + 1 + hi8) * 8 + colb] = (unsigned short)(p >> 16);
      }
    }
  }
}

// ---------------- im2col for 3x3 SAME conv on 6x9 image, K padded; rows >=54 zeroed
__device__ __forceinline__ void im2col3x3(const unsigned short* img, unsigned short* dst,
                                          int Ci, int Kpad){
  const int Kv = 9 * Ci;
  for (int e = threadIdx.x; e < 64 * Kpad; e += 256){
    int row = e / Kpad, k = e - row * Kpad;
    unsigned short v = 0;
    if (row < HWP && k < Kv){
      int tap = k / Ci, c = k - tap * Ci;
      int y = row / 9, x = row - y * 9;
      int sy = y + tap / 3 - 1, sx = x + (tap - (tap/3)*3) - 1;
      if (sy >= 0 && sy < HIMG && sx >= 0 && sx < WIMG) v = img[(sy*9 + sx)*Ci + c];
    }
    dst[e] = v;
  }
}

// ---------------- fused 3x3 conv stack (8->16->32->32), 2 images per workgroup
__global__ void k_conv3x3(const unsigned short* __restrict__ c3out,
                          const unsigned short* __restrict__ wt4, const float* __restrict__ b4,
                          const unsigned short* __restrict__ wt5, const float* __restrict__ b5,
                          const unsigned short* __restrict__ wt6, const float* __restrict__ b6,
                          unsigned short* __restrict__ c6out){
  extern __shared__ unsigned short smem[];
  unsigned short* Ab  = smem;                 // 2*64*KP6 (reused across layers)
  unsigned short* yin = Ab + 2*64*KP6;        // 2*54*8
  unsigned short* y4  = yin + 2*54*8;         // 2*54*16
  unsigned short* y5  = y4 + 2*54*16;         // 2*54*32
  const int tid = threadIdx.x, wave = tid >> 5, lane = tid & 31;
  const long long img0 = (long long)blockIdx.x * 2;
  for (int e = tid; e < 2*54*8; e += 256) yin[e] = c3out[img0*54*8 + e];
  __syncthreads();
  const int img = wave >> 2, rt = wave & 3;   // 2 images x 4 row-tiles = 8 waves
  const int colb = lane & 15, hi8 = (lane >> 4) << 3;
  // conv4: K=96 (72 valid), Co=16
  for (int im=0; im<2; im++) im2col3x3(yin + im*54*8, Ab + im*64*KP4, 8, KP4);
  __syncthreads();
  {
    v8f acc = gemm_tile(Ab + (img*64 + rt*16)*KP4, KP4, wt4, 96, 0, 96, zero8());
    const float bias = b4[colb];
    #pragma unroll
    for (int i=0;i<8;i++){
      int r = rt*16 + i + hi8;
      if (r < HWP) y4[img*54*16 + r*16 + colb] = f2bf(relu_(acc[i] + bias));
    }
  }
  __syncthreads();
  // conv5: K=160 (144 valid), Co=32
  for (int im=0; im<2; im++) im2col3x3(y4 + im*54*16, Ab + im*64*KP5, 16, KP5);
  __syncthreads();
  #pragma unroll
  for (int ct=0; ct<2; ct++){
    v8f acc = gemm_tile(Ab + (img*64 + rt*16)*KP5, KP5, wt5, 160, ct, 160, zero8());
    const int col = ct*16 + colb;
    const float bias = b5[col];
    #pragma unroll
    for (int i=0;i<8;i++){
      int r = rt*16 + i + hi8;
      if (r < HWP) y5[img*54*32 + r*32 + col] = f2bf(relu_(acc[i] + bias));
    }
  }
  __syncthreads();
  // conv6: K=288, Co=32
  for (int im=0; im<2; im++) im2col3x3(y5 + im*54*32, Ab + im*64*KP6, 32, KP6);
  __syncthreads();
  #pragma unroll
  for (int ct=0; ct<2; ct++){
    v8f acc = gemm_tile(Ab + (img*64 + rt*16)*KP6, KP6, wt6, 288, ct, 288, zero8());
    const int col = ct*16 + colb;
    const float bias = b6[col];
    #pragma unroll
    for (int i=0;i<8;i++){
      int r = rt*16 + i + hi8;
      if (r < HWP) c6out[((img0 + img)*54 + r)*32 + col] = f2bf(relu_(acc[i] + bias));
    }
  }
}

// ---------------- dense(1728->128) + relu + LayerNorm + 3 input projections
__global__ void k_dense(const unsigned short* __restrict__ act,
                        const unsigned short* __restrict__ wtD, const float* __restrict__ db,
                        const float* __restrict__ lns, const float* __restrict__ lnb,
                        const unsigned short* __restrict__ wtZ, const unsigned short* __restrict__ wtR,
                        const unsigned short* __restrict__ wtH,
                        float* __restrict__ wiz, float* __restrict__ wir, float* __restrict__ wih){
  extern __shared__ unsigned short smem[];
  const int wave = threadIdx.x >> 5, lane = threadIdx.x & 31;
  unsigned short* E = smem + wave * (16*SH);  // 16x128 bf16 emb tile (padded stride)
  const long long rowbase = (long long)blockIdx.x * 128 + wave * 16;
  const unsigned short* A = act + rowbase * 1728;
  v8f acc[8];
  #pragma unroll
  for (int ct=0; ct<8; ct++) acc[ct] = zero8();
  for (int k=0; k<1728; k+=32){               // A loaded once, shared across 8 col tiles
    v16u a = frag_u16(A, 1728, k);
    if (k + 64 < 1728) __builtin_prefetch(A + (lane & 15)*1728 + k + 64);
    #pragma unroll
    for (int ct=0; ct<8; ct++){
      v16u b = frag_u16(wtD + ct*16*1728, 1728, k);
      acc[ct] = wmma_bf16(a, b, acc[ct]);
    }
  }
  const int colb = lane & 15, hi8 = (lane >> 4) << 3;
  #pragma unroll
  for (int ct=0; ct<8; ct++){
    const float bias = db[ct*16 + colb];
    #pragma unroll
    for (int i=0;i<8;i++) acc[ct][i] = relu_(acc[ct][i] + bias);
  }
  // LayerNorm: per row i -> reduce across 8 col tiles then 16 lanes (halves stay independent)
  float mu[8], rs[8];
  #pragma unroll
  for (int i=0;i<8;i++){
    float s=0.f, q=0.f;
    #pragma unroll
    for (int ct=0; ct<8; ct++){ float v = acc[ct][i]; s += v; q += v*v; }
    #pragma unroll
    for (int m=1; m<16; m<<=1){ s += __shfl_xor(s, m, 32); q += __shfl_xor(q, m, 32); }
    float mean = s * (1.f/128.f);
    float var  = q * (1.f/128.f) - mean*mean;
    mu[i] = mean; rs[i] = rsqrtf(var + 1e-6f);
  }
  #pragma unroll
  for (int ct=0; ct<8; ct++){
    const int col = ct*16 + colb;
    const float g = lns[col], be = lnb[col];
    #pragma unroll
    for (int i=0;i<8;i+=2){
      unsigned int p = f2bf_pk((acc[ct][i]   - mu[i])   * rs[i]   * g + be,
                               (acc[ct][i+1] - mu[i+1]) * rs[i+1] * g + be);
      E[(i + hi8)*SH + col]     = (unsigned short)(p & 0xFFFFu);
      E[(i + 1 + hi8)*SH + col] = (unsigned short)(p >> 16);
    }
  }
  const unsigned short* wts[3] = {wtZ, wtR, wtH};
  float* outs[3] = {wiz, wir, wih};
  #pragma unroll
  for (int m=0; m<3; m++){
    #pragma unroll
    for (int ct=0; ct<8; ct++){
      v8f a2 = gemm_tile(E, SH, wts[m], 128, ct, 128, zero8());
      const int col = ct*16 + colb;
      #pragma unroll
      for (int i=0;i<8;i++) outs[m][(rowbase + i + hi8)*128 + col] = a2[i];
    }
  }
}

// ---------------- GRU scan: batch rows independent -> 16 WGs x 16 rows, loop T inside.
// f32 master h + bf16 mirror in LDS so A-fragments are two conflict-free ds_load_b128.
__global__ void k_gru(const float* __restrict__ wiz, const float* __restrict__ wir,
                      const float* __restrict__ wih, const int* __restrict__ dones,
                      const float* __restrict__ h0,
                      const unsigned short* __restrict__ wtHZ, const unsigned short* __restrict__ wtHR,
                      const unsigned short* __restrict__ wtHH,
                      const float* __restrict__ bz, const float* __restrict__ br,
                      const float* __restrict__ bh,
                      float* __restrict__ finalH, unsigned short* __restrict__ gruout){
  __shared__ float          hbuf[16*128];
  __shared__ unsigned short hbf [16*SH];
  __shared__ unsigned short rhbf[16*SH];
  const int tid = threadIdx.x, wave = tid >> 5, lane = tid & 31;
  const int rowbase = blockIdx.x * 16;
  for (int e = tid; e < 2048; e += 256){
    float v = h0[rowbase*128 + e];
    hbuf[e] = v; hbf[(e >> 7)*SH + (e & 127)] = f2bf(v);
  }
  __syncthreads();
  const int ct = wave, colb = lane & 15, hi8 = (lane >> 4) << 3;
  const int col = ct*16 + colb;
  const float bzc = bz[col], brc = br[col], bhc = bh[col];
  const unsigned short* BZ = wtHZ + ct*16*128;
  const unsigned short* BR = wtHR + ct*16*128;
  const unsigned short* BH = wtHH + ct*16*128;
  for (int t=0; t<TT; t++){
    for (int e = tid; e < 2048; e += 256)
      if (dones[t*NB + rowbase + (e >> 7)]){ hbuf[e] = 0.f; hbf[(e >> 7)*SH + (e & 127)] = 0; }
    __syncthreads();
    v8f az = zero8(), ar = zero8();
    for (int k=0; k<128; k+=32){
      v16u a = frag_u16(hbf, SH, k);
      az = wmma_bf16(a, frag_u16(BZ, 128, k), az);
      ar = wmma_bf16(a, frag_u16(BR, 128, k), ar);
    }
    const long long g = (long long)t*NB + rowbase;
    float zf[8], hf[8];
    #pragma unroll
    for (int i=0;i<8;i++){
      int r = i + hi8;
      float z  = sigm_(wiz[(g+r)*128 + col] + az[i] + bzc);
      float rr = sigm_(wir[(g+r)*128 + col] + ar[i] + brc);
      float h  = hbuf[r*128 + col];
      zf[i] = z; hf[i] = h;
      rhbf[r*SH + col] = f2bf(rr * h);
    }
    __syncthreads();
    v8f ah = zero8();
    for (int k=0; k<128; k+=32)
      ah = wmma_bf16(frag_u16(rhbf, SH, k), frag_u16(BH, 128, k), ah);
    #pragma unroll
    for (int i=0;i<8;i++){
      int r = i + hi8;
      float hh = tanhf(wih[(g+r)*128 + col] + ah[i] + bhc);
      float nh = (1.f - zf[i]) * hf[i] + zf[i] * hh;
      unsigned short nb = f2bf(nh);
      hbuf[r*128 + col] = nh;                      // per-wave disjoint columns
      hbf [r*SH  + col] = nb;
      gruout[(g+r)*128 + col] = nb;
    }
    __syncthreads();
  }
  for (int e = tid; e < 2048; e += 256) finalH[rowbase*128 + e] = hbuf[e];
}

// ---------------- actor & critic heads
__global__ void k_heads(const unsigned short* __restrict__ flat,
                        const unsigned short* __restrict__ wtAF, const float* __restrict__ afb,
                        const unsigned short* __restrict__ wtAO, const float* __restrict__ aob,
                        const unsigned short* __restrict__ wtCF, const float* __restrict__ cfb,
                        const unsigned short* __restrict__ wtCO, const float* __restrict__ cob,
                        float* __restrict__ logits, float* __restrict__ value){
  extern __shared__ unsigned short smem[];
  const int wave = threadIdx.x >> 5, lane = threadIdx.x & 31;
  unsigned short* Hb = smem + wave * (16*SH);
  const long long rowbase = (long long)blockIdx.x * 128 + wave * 16;
  const unsigned short* A = flat + rowbase * 128;
  const int colb = lane & 15, hi8 = (lane >> 4) << 3;
  #pragma unroll
  for (int ct=0; ct<8; ct++){
    v8f acc = gemm_tile(A, 128, wtAF, 128, ct, 128, zero8());
    const int col = ct*16 + colb;
    store_tile_relu_bf16(Hb, SH, col, acc, afb[col], hi8);
  }
  {
    v8f acc = gemm_tile(Hb, SH, wtAO, 128, 0, 128, zero8());
    if (colb < 6){
      const float bias = aob[colb];
      #pragma unroll
      for (int i=0;i<8;i++) logits[(rowbase + i + hi8)*6 + colb] = acc[i] + bias;
    }
  }
  #pragma unroll
  for (int ct=0; ct<8; ct++){
    v8f acc = gemm_tile(A, 128, wtCF, 128, ct, 128, zero8());
    const int col = ct*16 + colb;
    store_tile_relu_bf16(Hb, SH, col, acc, cfb[col], hi8);
  }
  {
    v8f acc = gemm_tile(Hb, SH, wtCO, 128, 0, 128, zero8());
    if (colb == 0){
      const float bias = cob[0];
      #pragma unroll
      for (int i=0;i<8;i++) value[rowbase + i + hi8] = acc[i] + bias;
    }
  }
}

extern "C" void kernel_launch(void* const* d_in, const int* in_sizes, int n_in,
                              void* d_out, int out_size, void* d_ws, size_t ws_size,
                              hipStream_t stream){
  (void)in_sizes; (void)n_in; (void)out_size; (void)ws_size;
  const float* obs  = (const float*)d_in[0];
  const float* h0   = (const float*)d_in[1];
  const int*   dones= (const int*)d_in[2];
  const float* c1w=(const float*)d_in[3],  *c1b=(const float*)d_in[4];
  const float* c2w=(const float*)d_in[5],  *c2b=(const float*)d_in[6];
  const float* c3w=(const float*)d_in[7],  *c3b=(const float*)d_in[8];
  const float* c4w=(const float*)d_in[9],  *c4b=(const float*)d_in[10];
  const float* c5w=(const float*)d_in[11], *c5b=(const float*)d_in[12];
  const float* c6w=(const float*)d_in[13], *c6b=(const float*)d_in[14];
  const float* dw =(const float*)d_in[15], *db =(const float*)d_in[16];
  const float* lns=(const float*)d_in[17], *lnb=(const float*)d_in[18];
  const float* wizk=(const float*)d_in[19],*wirk=(const float*)d_in[20],*wihk=(const float*)d_in[21];
  const float* whz=(const float*)d_in[22], *whr=(const float*)d_in[23], *whh=(const float*)d_in[24];
  const float* bz =(const float*)d_in[25], *br =(const float*)d_in[26], *bh =(const float*)d_in[27];
  const float* afw=(const float*)d_in[28], *afb=(const float*)d_in[29];
  const float* aow=(const float*)d_in[30], *aob=(const float*)d_in[31];
  const float* cfw=(const float*)d_in[32], *cfb=(const float*)d_in[33];
  const float* cow=(const float*)d_in[34], *cob=(const float*)d_in[35];

  float* out    = (float*)d_out;
  float* finalH = out;                 // 256*128
  float* logits = out + NB*HID;        // 64*256*6
  float* value  = logits + TN*6;       // 64*256

  char* ws = (char*)d_ws;
  size_t off = 0;
  auto take = [&](size_t bytes)->void*{
    size_t o = (off + 255) & ~(size_t)255;
    off = o + bytes;
    return (void*)(ws + o);
  };
  unsigned short* WT1 =(unsigned short*)take(128*32*2);
  unsigned short* WT2 =(unsigned short*)take(128*128*2);
  unsigned short* WT3 =(unsigned short*)take(16*128*2);
  unsigned short* WT4 =(unsigned short*)take(16*96*2);
  unsigned short* WT5 =(unsigned short*)take(32*160*2);
  unsigned short* WT6 =(unsigned short*)take(32*288*2);
  unsigned short* WTD =(unsigned short*)take((size_t)128*1728*2);
  unsigned short* WTZ =(unsigned short*)take(128*128*2);
  unsigned short* WTR =(unsigned short*)take(128*128*2);
  unsigned short* WTH =(unsigned short*)take(128*128*2);
  unsigned short* WTHZ=(unsigned short*)take(128*128*2);
  unsigned short* WTHR=(unsigned short*)take(128*128*2);
  unsigned short* WTHH=(unsigned short*)take(128*128*2);
  unsigned short* WTAF=(unsigned short*)take(128*128*2);
  unsigned short* WTCF=(unsigned short*)take(128*128*2);
  unsigned short* WTAO=(unsigned short*)take(16*128*2);
  unsigned short* WTCO=(unsigned short*)take(16*128*2);
  unsigned short* C3  =(unsigned short*)take((size_t)M1*8*2);
  unsigned short* C6  =(unsigned short*)take((size_t)M1*32*2);
  float* WIZ =(float*)take((size_t)TN*128*4);
  float* WIR =(float*)take((size_t)TN*128*4);
  float* WIH =(float*)take((size_t)TN*128*4);
  unsigned short* GRU =(unsigned short*)take((size_t)TN*128*2);

  auto pack = [&](const float* s, unsigned short* d, int Ks, int Ns, int Kp, int Np){
    int tot = Kp*Np;
    k_pack<<<(tot + 255)/256, 256, 0, stream>>>(s, d, Ks, Ns, Kp, Np);
  };
  pack(c1w, WT1, 26, 128, 32, 128);
  pack(c2w, WT2, 128, 128, 128, 128);
  pack(c3w, WT3, 128, 8, 128, 16);
  pack(c4w, WT4, 72, 16, 96, 16);
  pack(c5w, WT5, 144, 32, 160, 32);
  pack(c6w, WT6, 288, 32, 288, 32);
  pack(dw,  WTD, 1728, 128, 1728, 128);
  pack(wizk, WTZ, 128, 128, 128, 128);
  pack(wirk, WTR, 128, 128, 128, 128);
  pack(wihk, WTH, 128, 128, 128, 128);
  pack(whz, WTHZ, 128, 128, 128, 128);
  pack(whr, WTHR, 128, 128, 128, 128);
  pack(whh, WTHH, 128, 128, 128, 128);
  pack(afw, WTAF, 128, 128, 128, 128);
  pack(cfw, WTCF, 128, 128, 128, 128);
  pack(aow, WTAO, 128, 6, 128, 16);
  pack(cow, WTCO, 128, 1, 128, 16);

  k_conv1x1<<<M1/128, 256, 8*(16*SX + 2*16*SH)*2, stream>>>(obs, WT1, c1b, WT2, c2b, WT3, c3b, C3);
  k_conv3x3<<<TN/2, 256, (2*64*KP6 + 2*54*8 + 2*54*16 + 2*54*32)*2, stream>>>(
      C3, WT4, c4b, WT5, c5b, WT6, c6b, C6);
  k_dense<<<TN/128, 256, 8*(16*SH)*2, stream>>>(C6, WTD, db, lns, lnb, WTZ, WTR, WTH,
                                                WIZ, WIR, WIH);
  k_gru<<<NB/16, 256, 0, stream>>>(WIZ, WIR, WIH, dones, h0, WTHZ, WTHR, WTHH,
                                   bz, br, bh, finalH, GRU);
  k_heads<<<TN/128, 256, 8*(16*SH)*2, stream>>>(GRU, WTAF, afb, WTAO, aob, WTCF, cfb, WTCO, cob,
                                                logits, value);
}
